// ResGCN_77129022701586
// MI455X (gfx1250) — matrix operand order repaired and checked by reading, not compiled, not run on hardware
//
#include <hip/hip_runtime.h>
#include <hip/hip_bf16.h>

typedef float    v4f  __attribute__((ext_vector_type(4)));
typedef _Float16 v4h  __attribute__((ext_vector_type(4)));
typedef _Float16 v8h  __attribute__((ext_vector_type(8)));
typedef _Float16 v16h __attribute__((ext_vector_type(16)));
typedef float    v8f  __attribute__((ext_vector_type(8)));

__device__ __forceinline__ v16h cat8(v8h lo, v8h hi) {
  return __builtin_shufflevector(lo, hi, 0,1,2,3,4,5,6,7,8,9,10,11,12,13,14,15);
}
__device__ __forceinline__ v8f wmma16(v16h a, v16h b, v8f c) {
  // D = A(16x32 f16) x B(32x16 f16) + C(16x16 f32)
  return __builtin_amdgcn_wmma_f32_16x16x32_f16(false, a, false, b, (short)0, c, false, false);
}

// ---------------------------------------------------------------------------
// Generic NN GEMM: C[M,N] = post * (A[M,K] @ B[K,N]) (+Cadd) (+bias) (relu?)
// post = alpha (or alpha / *scale_sum).  A,B fp32 in global, f16 in LDS.
// 128 threads = 4 waves; block tile 64x64; wave tile 32x32 (2x2 WMMA).
// ALIGNED=true: M%64==0, N%64==0, K%32==0, lda%4==0, ldb%4==0 -> unguarded
// vectorized staging (b128 global loads, b64 LDS stores) + K-tile prefetch.
// ---------------------------------------------------------------------------
template <bool ALIGNED>
__global__ __launch_bounds__(128)
void gemm_nn_f16w(const float* __restrict__ A, int lda,
                  const float* __restrict__ B, int ldb,
                  float* __restrict__ C, int ldc,
                  int M, int N, int K,
                  const float* __restrict__ bias,
                  const float* __restrict__ scale_sum, float alpha,
                  const float* __restrict__ Cadd, int ldadd,
                  int relu)
{
  __shared__ __align__(16) _Float16 As[64][40];
  __shared__ __align__(16) _Float16 Bs[64][40];

  const int tid  = threadIdx.x;
  const int lane = tid & 31;
  const int wave = tid >> 5;
  const int wr = wave >> 1, wc = wave & 1;
  const int tileM = blockIdx.y * 64, tileN = blockIdx.x * 64;

  v8f acc00 = {}, acc01 = {}, acc10 = {}, acc11 = {};

  const int lm   = lane & 15;
  const int ksel = lane >> 4;      // which half of the wave
  const int ah   = ksel * 8;       // A: half-offset within 32-K chunk
  const int bh   = ksel * 16;      // unused: B loads use ah, same k order as A

  for (int k0 = 0; k0 < K; k0 += 32) {
    __syncthreads();
    if (ALIGNED) {
      // Stage A: 2048 floats, 4 x b128 per thread, b64 LDS stores
#pragma unroll
      for (int i = 0; i < 4; ++i) {
        int e4 = tid + i * 128;
        int r = e4 >> 3, c4 = (e4 & 7) * 4;
        v4f f = *(const v4f*)(A + (long)(tileM + r) * lda + k0 + c4);
        *(v4h*)&As[r][c4] = __builtin_convertvector(f, v4h);
      }
      // Stage B (K x N row-major -> transposed [n][k])
#pragma unroll
      for (int i = 0; i < 4; ++i) {
        int e4 = tid + i * 128;
        int c = e4 >> 4, n4 = (e4 & 15) * 4;
        v4f f = *(const v4f*)(B + (long)(k0 + c) * ldb + tileN + n4);
        Bs[n4 + 0][c] = (_Float16)f[0];
        Bs[n4 + 1][c] = (_Float16)f[1];
        Bs[n4 + 2][c] = (_Float16)f[2];
        Bs[n4 + 3][c] = (_Float16)f[3];
      }
      if (k0 + 32 < K) {  // prefetch next K-tile
        int r = tid >> 1;
        __builtin_prefetch(A + (long)(tileM + r) * lda + (k0 + 32) + (tid & 1) * 16, 0, 0);
        __builtin_prefetch(B + (long)(k0 + 32 + (tid >> 2)) * ldb + tileN + (tid & 3) * 16, 0, 0);
      }
    } else {
      for (int e = tid; e < 64 * 32; e += 128) {
        int r = e >> 5, c = e & 31;
        int gm = tileM + r, gk = k0 + c;
        float v = (gm < M && gk < K) ? A[(long)gm * lda + gk] : 0.f;
        As[r][c] = (_Float16)v;
      }
      for (int e = tid; e < 64 * 32; e += 128) {
        int c = e >> 6, n = e & 63;
        int gk = k0 + c, gn = tileN + n;
        float v = (gk < K && gn < N) ? B[(long)gk * ldb + gn] : 0.f;
        Bs[n][c] = (_Float16)v;
      }
    }
    __syncthreads();

    const _Float16* a0p = &As[wr * 32 + lm][0];
    const _Float16* a1p = &As[wr * 32 + 16 + lm][0];
    v16h a0 = cat8(*(const v8h*)(a0p + ah), *(const v8h*)(a0p + 16 + ah));
    v16h a1 = cat8(*(const v8h*)(a1p + ah), *(const v8h*)(a1p + 16 + ah));
    const _Float16* b0p = &Bs[wc * 32 + lm][0];
    const _Float16* b1p = &Bs[wc * 32 + 16 + lm][0];
    v16h b0 = cat8(*(const v8h*)(b0p + ah), *(const v8h*)(b0p + 16 + ah));
    v16h b1 = cat8(*(const v8h*)(b1p + ah), *(const v8h*)(b1p + 16 + ah));

    acc00 = wmma16(a0, b0, acc00);
    acc01 = wmma16(a0, b1, acc01);
    acc10 = wmma16(a1, b0, acc10);
    acc11 = wmma16(a1, b1, acc11);
  }

  float post = alpha;
  if (scale_sum) post = alpha / scale_sum[0];

  const int mrow0 = ksel * 8;
  auto store_tile = [&](v8f acc, int rt, int ct) {
    int gn = tileN + wc * 32 + ct * 16 + lm;
    if (!ALIGNED && gn >= N) return;
    int mbase = tileM + wr * 32 + rt * 16 + mrow0;
#pragma unroll
    for (int i = 0; i < 8; ++i) {
      int gm = mbase + i;
      if (!ALIGNED && gm >= M) continue;
      float v = acc[i] * post;
      if (Cadd) v += Cadd[(long)gm * ldadd + gn];
      if (bias) v += bias[gn];
      if (relu) v = fmaxf(v, 0.f);
      C[(long)gm * ldc + gn] = v;
    }
  };
  store_tile(acc00, 0, 0);
  store_tile(acc01, 0, 1);
  store_tile(acc10, 1, 0);
  store_tile(acc11, 1, 1);
}

// ---------------------------------------------------------------------------
// Fused attention scores (NT x 2):  S = XK@XQ^T/16,  T = G@G^T/16,
// Mout = relu(T)*exp(S), and global sum via atomicAdd.  4096x4096, K=256.
// NT layout: B tiles stage directly into [n][k] LDS layout, all vectorized.
// ---------------------------------------------------------------------------
__global__ __launch_bounds__(128)
void attn_scores_f16w(const float* __restrict__ XK,
                      const float* __restrict__ XQ,
                      const float* __restrict__ G,
                      float* __restrict__ Mout,
                      float* __restrict__ sum_ptr)
{
  __shared__ __align__(16) _Float16 Ak[64][40];
  __shared__ __align__(16) _Float16 Bq[64][40];
  __shared__ __align__(16) _Float16 Ag[64][40];
  __shared__ __align__(16) _Float16 Bg[64][40];
  __shared__ float red[128];

  const int tid  = threadIdx.x;
  const int lane = tid & 31;
  const int wave = tid >> 5;
  const int wr = wave >> 1, wc = wave & 1;
  const int tileM = blockIdx.y * 64, tileN = blockIdx.x * 64;

  v8f s00 = {}, s01 = {}, s10 = {}, s11 = {};
  v8f g00 = {}, g01 = {}, g10 = {}, g11 = {};

  const int lm   = lane & 15;
  const int ksel = lane >> 4;
  const int ah   = ksel * 8;
  const int bh   = ksel * 16;

  for (int k0 = 0; k0 < 256; k0 += 32) {
    __syncthreads();
#pragma unroll
    for (int i = 0; i < 4; ++i) {
      int e4 = tid + i * 128;
      int r = e4 >> 3, c4 = (e4 & 7) * 4;
      long offM = (long)(tileM + r) * 256 + k0 + c4;
      long offN = (long)(tileN + r) * 256 + k0 + c4;
      *(v4h*)&Ak[r][c4] = __builtin_convertvector(*(const v4f*)(XK + offM), v4h);
      *(v4h*)&Bq[r][c4] = __builtin_convertvector(*(const v4f*)(XQ + offN), v4h);
      *(v4h*)&Ag[r][c4] = __builtin_convertvector(*(const v4f*)(G + offM), v4h);
      *(v4h*)&Bg[r][c4] = __builtin_convertvector(*(const v4f*)(G + offN), v4h);
    }
    __syncthreads();

    const _Float16* ak0 = &Ak[wr * 32 + lm][0];
    const _Float16* ak1 = &Ak[wr * 32 + 16 + lm][0];
    const _Float16* ag0 = &Ag[wr * 32 + lm][0];
    const _Float16* ag1 = &Ag[wr * 32 + 16 + lm][0];
    v16h aK0 = cat8(*(const v8h*)(ak0 + ah), *(const v8h*)(ak0 + 16 + ah));
    v16h aK1 = cat8(*(const v8h*)(ak1 + ah), *(const v8h*)(ak1 + 16 + ah));
    v16h aG0 = cat8(*(const v8h*)(ag0 + ah), *(const v8h*)(ag0 + 16 + ah));
    v16h aG1 = cat8(*(const v8h*)(ag1 + ah), *(const v8h*)(ag1 + 16 + ah));

    const _Float16* bq0 = &Bq[wc * 32 + lm][0];
    const _Float16* bq1 = &Bq[wc * 32 + 16 + lm][0];
    const _Float16* bg0 = &Bg[wc * 32 + lm][0];
    const _Float16* bg1 = &Bg[wc * 32 + 16 + lm][0];
    v16h bQ0 = cat8(*(const v8h*)(bq0 + ah), *(const v8h*)(bq0 + 16 + ah));
    v16h bQ1 = cat8(*(const v8h*)(bq1 + ah), *(const v8h*)(bq1 + 16 + ah));
    v16h bG0 = cat8(*(const v8h*)(bg0 + ah), *(const v8h*)(bg0 + 16 + ah));
    v16h bG1 = cat8(*(const v8h*)(bg1 + ah), *(const v8h*)(bg1 + 16 + ah));

    s00 = wmma16(aK0, bQ0, s00);
    s01 = wmma16(aK0, bQ1, s01);
    s10 = wmma16(aK1, bQ0, s10);
    s11 = wmma16(aK1, bQ1, s11);
    g00 = wmma16(aG0, bG0, g00);
    g01 = wmma16(aG0, bG1, g01);
    g10 = wmma16(aG1, bG0, g10);
    g11 = wmma16(aG1, bG1, g11);
  }

  float lsum = 0.f;
  const int mrow0 = ksel * 8;
  auto emit = [&](v8f s, v8f g, int rt, int ct) {
    int gn = tileN + wc * 32 + ct * 16 + lm;
    int mbase = tileM + wr * 32 + rt * 16 + mrow0;
#pragma unroll
    for (int i = 0; i < 8; ++i) {
      float sv = s[i] * 0.0625f;
      float gv = fmaxf(g[i] * 0.0625f, 0.f);
      float m  = gv * __expf(sv);
      Mout[(long)(mbase + i) * 4096 + gn] = m;
      lsum += m;
    }
  };
  emit(s00, g00, 0, 0);
  emit(s01, g01, 0, 1);
  emit(s10, g10, 1, 0);
  emit(s11, g11, 1, 1);

  red[tid] = lsum;
  __syncthreads();
  for (int s = 64; s >= 1; s >>= 1) {
    if (tid < s) red[tid] += red[tid + s];
    __syncthreads();
  }
  if (tid == 0) atomicAdd(sum_ptr, red[0]);
}

// ---------------------------------------------------------------------------
// Small helper kernels
// ---------------------------------------------------------------------------
__global__ void zero_kernel(float* p, int n) {
  int i = blockIdx.x * blockDim.x + threadIdx.x;
  if (i < n) p[i] = 0.f;
}

__global__ void gather_kernel(const int* __restrict__ subobj,
                              const float* __restrict__ tmatrix,
                              const float* __restrict__ tmatbin,
                              const float* __restrict__ twords,
                              float* __restrict__ m_sub_out,
                              float* __restrict__ mbin,
                              float* __restrict__ o1w,
                              float* __restrict__ o2w)
{
  int r = blockIdx.x;
  int sub = subobj[r * 2 + 0], obj = subobj[r * 2 + 1];
  long base = (long)sub * 151 + obj;
  for (int c = threadIdx.x; c < 51; c += blockDim.x)
    m_sub_out[(long)r * 51 + c] = tmatrix[base * 51 + c];
  for (int c = threadIdx.x; c < 200; c += blockDim.x) {
    o1w[(long)r * 200 + c] = twords[(long)(sub - 1) * 200 + c];
    o2w[(long)r * 200 + c] = twords[(long)(obj - 1) * 200 + c];
  }
  if (threadIdx.x == 0) mbin[r] = tmatbin[base * 4 + 1];  // [sub,obj,0,1]
}

__global__ void build_cat912_kernel(const float* __restrict__ o1w,
                                    const float* __restrict__ x0,
                                    const float* __restrict__ o2w,
                                    float* __restrict__ cat)
{
  long i = (long)blockIdx.x * blockDim.x + threadIdx.x;
  if (i >= (long)4096 * 912) return;
  int r = (int)(i / 912), c = (int)(i % 912);
  float v;
  if (c < 200)       v = o1w[(long)r * 200 + c];
  else if (c < 712)  v = x0[(long)r * 512 + (c - 200)];
  else               v = o2w[(long)r * 200 + (c - 712)];
  cat[i] = v;
}

__global__ void build_cat712_kernel(const float* __restrict__ ggnn, int featOff,
                                    const float* __restrict__ wsrc,
                                    float* __restrict__ cat)
{
  long i = (long)blockIdx.x * blockDim.x + threadIdx.x;
  if (i >= (long)4096 * 712) return;
  int r = (int)(i / 712), c = (int)(i % 712);
  float v = (c < 512) ? ggnn[(long)r * 1536 + featOff + c]
                      : wsrc[(long)r * 200 + (c - 512)];
  cat[i] = v;
}

__global__ void bin_kernel(const float* __restrict__ xfin,
                           const float* __restrict__ w,
                           const float* __restrict__ b,
                           const float* __restrict__ mbin,
                           float* __restrict__ out1)
{
  int r = blockIdx.x * blockDim.x + threadIdx.x;
  if (r >= 4096) return;
  float acc = b[0] + mbin[r];
  const float* xr = xfin + (long)r * 512;
#pragma unroll 8
  for (int k = 0; k < 512; ++k) acc += xr[k] * w[k];
  out1[r] = acc;
}

__global__ void combine_kernel(const float* __restrict__ r1,
                               const float* __restrict__ msub,
                               const float* __restrict__ d1,
                               const float* __restrict__ d2,
                               const float* __restrict__ bx,
                               float* __restrict__ out0)
{
  long i = (long)blockIdx.x * blockDim.x + threadIdx.x;
  if (i >= (long)4096 * 51) return;
  out0[i] = r1[i] + 5.f * msub[i] + d1[i] + d2[i] + bx[i];
}

// ---------------------------------------------------------------------------
// Host orchestration
// ---------------------------------------------------------------------------
extern "C" void kernel_launch(void* const* d_in, const int* in_sizes, int n_in,
                              void* d_out, int out_size, void* d_ws, size_t ws_size,
                              hipStream_t stream)
{
  (void)in_sizes; (void)n_in; (void)out_size; (void)ws_size;

  // JAX pytree (sorted-key) flatten order of setup_inputs():
  const float* box      = (const float*)d_in[0];   // box_features (4096,22)
  const float* ggnn     = (const float*)d_in[1];   // input_ggnn (4096,1,1536)
  const float* bin_b    = (const float*)d_in[2];   // params.bin_cls.b (1)
  const float* bin_w    = (const float*)d_in[3];   // params.bin_cls.w (512,1)
  const float* WG1      = (const float*)d_in[4];   // blocks.WG1 (9,22,256)
  const float* WG2      = (const float*)d_in[5];
  const float* WK1      = (const float*)d_in[6];   // (9,512,256)
  const float* WK2      = (const float*)d_in[7];
  const float* WQ1      = (const float*)d_in[8];
  const float* WQ2      = (const float*)d_in[9];
  const float* WV1      = (const float*)d_in[10];
  const float* WV2      = (const float*)d_in[11];
  const float* box_b    = (const float*)d_in[12];  // (64)
  const float* box_w    = (const float*)d_in[13];  // (22,64)
  const float* boxcls_b = (const float*)d_in[14];  // (51)
  const float* boxcls_w = (const float*)d_in[15];  // (64,51)
  const float* cls_b    = (const float*)d_in[16];  // (51)
  const float* cls_w    = (const float*)d_in[17];  // (512,51)
  const float* emd1_b   = (const float*)d_in[18];  // (512)
  const float* emd1_w   = (const float*)d_in[19];  // (1536,512)
  const float* emd2_b   = (const float*)d_in[20];  // (512)
  const float* emd2_w   = (const float*)d_in[21];  // (912,512)
  const float* o1_b     = (const float*)d_in[22];  // (512)
  const float* o1_w     = (const float*)d_in[23];  // (712,512)
  const float* o1cls_b  = (const float*)d_in[24];  // (51)
  const float* o1cls_w  = (const float*)d_in[25];  // (512,51)
  const float* o2_b     = (const float*)d_in[26];
  const float* o2_w     = (const float*)d_in[27];
  const float* o2cls_b  = (const float*)d_in[28];
  const float* o2cls_w  = (const float*)d_in[29];
  /* d_in[30] = rel_inds (unused) */
  const int*   subobj   = (const int*)d_in[31];    // sub_obj_preds (4096,2)
  const float* tmatrix  = (const float*)d_in[32];  // (151,151,51)
  const float* tmatbin  = (const float*)d_in[33];  // (151,151,2,2)
  const float* twords   = (const float*)d_in[34];  // (150,200)

  // Workspace layout (floats)
  float* W = (float*)d_ws;
  float* sum  = W;                              // 64
  float* mbin = W + 64;                         // 4096
  float* o1w  = W + 64 + 4096;                  // 4096*200
  float* o2w  = o1w + (long)4096 * 200;
  float* xA   = o2w + (long)4096 * 200;         // 4096*512
  float* xB   = xA + (long)4096 * 512;
  float* XK   = xB + (long)4096 * 512;          // 4096*256 each
  float* XQ   = XK + (long)4096 * 256;
  float* XV   = XQ + (long)4096 * 256;
  float* Gm   = XV + (long)4096 * 256;
  float* BIG  = Gm + (long)4096 * 256;          // 4096*4096 reused region
  float* x0     = BIG;                          // early: 4096*512
  float* cat912 = BIG + (long)4096 * 512;       // early: 4096*912
  float* Mbuf   = BIG;                          // blocks: 4096*4096
  float* cat712 = BIG;                          // heads: 4096*712
  float* hid    = BIG + (long)4096 * 712;       // heads: 4096*512
  float* bhid   = hid + (long)4096 * 512;       // heads: 4096*64

  float* out0 = (float*)d_out;                  // rel_dists (4096,51)
  float* out1 = out0 + (long)4096 * 51;         // rel_binary (4096,1)
  float* out2 = out1 + 4096;                    // rel_dists1
  float* out3 = out2 + (long)4096 * 51;         // m_sub
  float* out4 = out3 + (long)4096 * 51;         // o1_dists
  float* out5 = out4 + (long)4096 * 51;         // o2_dists
  float* out6 = out5 + (long)4096 * 51;         // box_dist

  auto gemm = [&](const float* A, int lda, const float* B, int ldb,
                  float* C, int ldc, int M, int N, int K,
                  const float* bias, const float* ssum, float alpha,
                  const float* Cadd, int ldadd, int relu) {
    dim3 g((N + 63) / 64, (M + 63) / 64), b(128);
    bool al = (M % 64 == 0) && (N % 64 == 0) && (K % 32 == 0) &&
              (lda % 4 == 0) && (ldb % 4 == 0);
    if (al)
      gemm_nn_f16w<true><<<g, b, 0, stream>>>(A, lda, B, ldb, C, ldc, M, N, K,
                                              bias, ssum, alpha, Cadd, ldadd, relu);
    else
      gemm_nn_f16w<false><<<g, b, 0, stream>>>(A, lda, B, ldb, C, ldc, M, N, K,
                                               bias, ssum, alpha, Cadd, ldadd, relu);
  };

  // 1) gathers: m_sub -> out3, mbin, word embeddings
  gather_kernel<<<dim3(4096), dim3(256), 0, stream>>>(
      subobj, tmatrix, tmatbin, twords, out3, mbin, o1w, o2w);

  // 2) emd1: x0 = ggnn @ W + b        [aligned path]
  gemm(ggnn, 1536, emd1_w, 512, x0, 512, 4096, 512, 1536, emd1_b, nullptr, 1.f, nullptr, 0, 0);
  // 3) cat912 = [o1w | x0 | o2w]
  build_cat912_kernel<<<dim3((4096 * 912 + 255) / 256), dim3(256), 0, stream>>>(
      o1w, x0, o2w, cat912);
  // 4) xA = relu(cat912 @ emd2 + b)   [generic: K=912]
  gemm(cat912, 912, emd2_w, 512, xA, 512, 4096, 512, 912, emd2_b, nullptr, 1.f, nullptr, 0, 1);

  // 5) 9 residual attention blocks (ping-pong xA <-> xB)
  for (int blk = 0; blk < 9; ++blk) {
    float* xin  = (blk & 1) ? xB : xA;
    float* xout = (blk & 1) ? xA : xB;
    for (int br = 0; br < 2; ++br) {
      const float* wk = (br == 0 ? WK1 : WK2) + (long)blk * 512 * 256;
      const float* wq = (br == 0 ? WQ1 : WQ2) + (long)blk * 512 * 256;
      const float* wv = (br == 0 ? WV1 : WV2) + (long)blk * 512 * 256;
      const float* wg = (br == 0 ? WG1 : WG2) + (long)blk * 22 * 256;
      gemm(xin, 512, wk, 256, XK, 256, 4096, 256, 512, nullptr, nullptr, 1.f, nullptr, 0, 0);
      gemm(xin, 512, wq, 256, XQ, 256, 4096, 256, 512, nullptr, nullptr, 1.f, nullptr, 0, 0);
      gemm(xin, 512, wv, 256, XV, 256, 4096, 256, 512, nullptr, nullptr, 1.f, nullptr, 0, 0);
      gemm(box, 22, wg, 256, Gm, 256, 4096, 256, 22, nullptr, nullptr, 1.f, nullptr, 0, 0);
      zero_kernel<<<dim3(1), dim3(64), 0, stream>>>(sum, 64);
      attn_scores_f16w<<<dim3(64, 64), dim3(128), 0, stream>>>(XK, XQ, Gm, Mbuf, sum);
      // xout[:, br*256:(br+1)*256] = 0.1/sum * (M @ XV) + xin[:, same]  [aligned]
      gemm(Mbuf, 4096, XV, 256, xout + br * 256, 512, 4096, 256, 4096,
           nullptr, sum, 0.1f, xin + br * 256, 512, 0);
    }
  }
  float* xfin = xB;  // 9 blocks: ends in xB

  // 6) heads
  gemm(xfin, 512, cls_w, 51, out2, 51, 4096, 51, 512, cls_b, nullptr, 1.f, nullptr, 0, 0);
  bin_kernel<<<dim3(16), dim3(256), 0, stream>>>(xfin, bin_w, bin_b, mbin, out1);

  build_cat712_kernel<<<dim3((4096 * 712 + 255) / 256), dim3(256), 0, stream>>>(
      ggnn, 0, o1w, cat712);
  gemm(cat712, 712, o1_w, 512, hid, 512, 4096, 512, 712, o1_b, nullptr, 1.f, nullptr, 0, 1);
  gemm(hid, 512, o1cls_w, 51, out4, 51, 4096, 51, 512, o1cls_b, nullptr, 1.f, nullptr, 0, 0);

  build_cat712_kernel<<<dim3((4096 * 712 + 255) / 256), dim3(256), 0, stream>>>(
      ggnn, 512, o2w, cat712);
  gemm(cat712, 712, o2_w, 512, hid, 512, 4096, 512, 712, o2_b, nullptr, 1.f, nullptr, 0, 1);
  gemm(hid, 512, o2cls_w, 51, out5, 51, 4096, 51, 512, o2cls_b, nullptr, 1.f, nullptr, 0, 0);

  gemm(box, 22, box_w, 64, bhid, 64, 4096, 64, 22, box_b, nullptr, 1.f, nullptr, 0, 1);
  gemm(bhid, 64, boxcls_w, 51, out6, 51, 4096, 51, 64, boxcls_b, nullptr, 1.f, nullptr, 0, 0);

  combine_kernel<<<dim3((4096 * 51 + 255) / 256), dim3(256), 0, stream>>>(
      out2, out3, out4, out5, out6, out0);
}
